// CSSGT_91036126806134
// MI455X (gfx1250) — matrix operand order repaired and checked by reading, not compiled, not run on hardware
//
#include <hip/hip_runtime.h>

#define Hdim 64
#define THv 0.7f
#define ATT_THv 0.2f
#define EPSv 1e-5f

typedef __attribute__((ext_vector_type(16))) __bf16 v16bf;
typedef __attribute__((ext_vector_type(8)))  float  v8f;

__device__ __forceinline__ unsigned short f2bf(float f) {
    union { float f; unsigned u; } cv; cv.f = f;
    unsigned u = cv.u;
    u += 0x7fffu + ((u >> 16) & 1u);   // round-to-nearest-even
    return (unsigned short)(u >> 16);
}

// pack two f32 -> {bf16(f1),bf16(f0)} in one VGPR via v_perm_b32
__device__ __forceinline__ unsigned pack_bf16(float f0, float f1) {
    union { float f; unsigned u; } a, b; a.f = f0; b.f = f1;
    unsigned u0 = a.u + (0x7fffu + ((a.u >> 16) & 1u));
    unsigned u1 = b.u + (0x7fffu + ((b.u >> 16) & 1u));
    return __builtin_amdgcn_perm(u1, u0, 0x07060302u);  // {u1[31:16], u0[31:16]}
}

// ---------------------------------------------------------------- GEMM (WMMA)
// C[N,64] = A[N,64] @ W[64,64], optional spike(A) transform on load.
// Block = 256 threads (8 waves); each wave computes a 16x64 output strip.
__global__ __launch_bounds__(256)
void gemm_wmma_kernel(const float* __restrict__ A, const float* __restrict__ Wf,
                      float* __restrict__ C, int spikeA, int nrows) {
    __shared__ unsigned short Wt[Hdim * Hdim];      // Wt[n*64+k] = bf16(W[k][n])
    const int tid = threadIdx.x;
    for (int i = tid; i < Hdim * Hdim; i += 256) {
        int k = i >> 6, n = i & 63;
        Wt[n * Hdim + k] = f2bf(Wf[i]);
    }
    __syncthreads();

    const int wave  = tid >> 5;
    const int lane  = tid & 31;
    const int mhalf = lane >> 4;        // which 16-lane half
    const int l15   = lane & 15;
    const int rowbase = blockIdx.x * 128 + wave * 16;   // wave-uniform

    union Frag { v16bf v; unsigned w[8]; };

    // A fragments (16x32 bf16) for K-steps 0 and 1
    Frag a[2];
    {
        int r = rowbase + l15;
        if (r >= nrows) r = nrows - 1;              // clamp; stores are guarded
        const float* arow = A + (unsigned)(r * Hdim);
        const int kofs = mhalf * 8;
#pragma unroll
        for (int ks = 0; ks < 2; ++ks) {
#pragma unroll
            for (int i = 0; i < 8; ++i) {
                const int k = ks * 32 + kofs + ((i < 4) ? (2 * i) : (8 + 2 * i));
                float2 f = *(const float2*)(arow + k);
                float f0 = f.x, f1 = f.y;
                if (spikeA) {
                    f0 = (f0 >= THv) ? 1.0f : 0.0f;
                    f1 = (f1 >= THv) ? 1.0f : 0.0f;
                }
                a[ks].w[i] = pack_bf16(f0, f1);
            }
        }
    }

    const bool fullTile = (rowbase + 16 <= nrows);   // wave-uniform branch

#pragma unroll
    for (int nt = 0; nt < 4; ++nt) {
        Frag b[2];
        const int n = nt * 16 + l15;
#pragma unroll
        for (int ks = 0; ks < 2; ++ks) {
            const int kb = ks * 32 + mhalf * 8;
#pragma unroll
            for (int j = 0; j < 8; ++j)              // pair (K=kb+o, kb+o+1), o = j<4 ? 2j : 8+2j
                b[ks].w[j] = *(const unsigned*)&Wt[n * Hdim + kb + ((j < 4) ? (2 * j) : (8 + 2 * j))];
        }
        v8f c = {};
        c = __builtin_amdgcn_wmma_f32_16x16x32_bf16(false, a[0].v, false, b[0].v,
                                                    (short)0, c, false, false);
        c = __builtin_amdgcn_wmma_f32_16x16x32_bf16(false, a[1].v, false, b[1].v,
                                                    (short)0, c, false, false);
        const int col = nt * 16 + l15;
        if (fullTile) {
#pragma unroll
            for (int r = 0; r < 8; ++r)
                C[(unsigned)((rowbase + r + mhalf * 8) * Hdim + col)] = c[r];
        } else {
#pragma unroll
            for (int r = 0; r < 8; ++r) {
                const int row = rowbase + r + mhalf * 8;
                if (row < nrows) C[(unsigned)(row * Hdim + col)] = c[r];
            }
        }
    }
}

// ---------------------------------------------------------------- small utils
__global__ void fill_kernel(float* __restrict__ p, float v, int n) {
    int i = blockIdx.x * blockDim.x + threadIdx.x;
    if (i < n) p[i] = v;
}

__global__ void deg_kernel(float* __restrict__ deg, const int* __restrict__ dst,
                           const int* __restrict__ mask, int nedges) {
    int e = blockIdx.x * blockDim.x + threadIdx.x;
    if (e >= nedges) return;
    float w = mask ? (float)mask[e] : 1.0f;
    if (w != 0.0f) atomicAdd(&deg[dst[e]], w);
}

__global__ void rsqrt_kernel(float* __restrict__ d, int n) {
    int i = blockIdx.x * blockDim.x + threadIdx.x;
    if (i < n) d[i] = rsqrtf(d[i]);
}

// agg[n,c] = dis[n]^2 * xw[n,c] + bias[c]   (self-loop + bias, init for scatter)
__global__ __launch_bounds__(256)
void agg_init_kernel(const float* __restrict__ xw, const float* __restrict__ dis,
                     const float* __restrict__ bias, float* __restrict__ agg, int nrows) {
    int tid = blockIdx.x * blockDim.x + threadIdx.x;
    if (tid >= nrows * (Hdim / 4)) return;
    const int row = tid >> 4;
    const int c   = (tid & 15) * 4;
    const float d2 = dis[row] * dis[row];
    const unsigned off = (unsigned)tid * 4;
    float4 xv = *(const float4*)(xw + off);
    float4 o;
    o.x = d2 * xv.x + bias[c + 0];
    o.y = d2 * xv.y + bias[c + 1];
    o.z = d2 * xv.z + bias[c + 2];
    o.w = d2 * xv.w + bias[c + 3];
    *(float4*)(agg + off) = o;
}

// scatter: agg[dst] += dis[src]*w*dis[dst] * xw[src]; 16 threads/edge, float4 each
__global__ __launch_bounds__(256)
void edge_agg_kernel(const float* __restrict__ xw, float* __restrict__ agg,
                     const int* __restrict__ src, const int* __restrict__ dst,
                     const int* __restrict__ mask, const float* __restrict__ dis,
                     int nedges) {
    int tid = blockIdx.x * blockDim.x + threadIdx.x;
    int e = tid >> 4;
    if (e >= nedges) return;
    float w = 1.0f;
    if (mask) { w = (float)mask[e]; if (w == 0.0f) return; }  // dropped edge: no atomics
    const int c = (tid & 15) * 4;
    const int s = src[e], d = dst[e];
    const float norm = dis[s] * w * dis[d];
    float4 xv = *(const float4*)(xw + (unsigned)(s * Hdim + c));
    float* ap = agg + (unsigned)(d * Hdim + c);
    atomicAdd(ap + 0, norm * xv.x);
    atomicAdd(ap + 1, norm * xv.y);
    atomicAdd(ap + 2, norm * xv.z);
    atomicAdd(ap + 3, norm * xv.w);
}

__global__ void zero_stats_kernel(float* __restrict__ stats) {
    if (threadIdx.x < 2 * Hdim) stats[threadIdx.x] = 0.0f;
}

// column sum / sumsq over N rows
__global__ __launch_bounds__(256)
void stats_kernel(const float* __restrict__ x, float* __restrict__ stats, int nrows) {
    const int c  = threadIdx.x & 63;
    const int rg = threadIdx.x >> 6;        // 0..3
    float s = 0.f, sq = 0.f;
    for (int r = blockIdx.x * 4 + rg; r < nrows; r += gridDim.x * 4) {
        float v = x[(unsigned)(r * Hdim + c)];
        s += v; sq += v * v;
    }
    __shared__ float ls[4][Hdim], lq[4][Hdim];
    ls[rg][c] = s; lq[rg][c] = sq;
    __syncthreads();
    if (rg == 0) {
        s  = ls[0][c] + ls[1][c] + ls[2][c] + ls[3][c];
        sq = lq[0][c] + lq[1][c] + lq[2][c] + lq[3][c];
        atomicAdd(&stats[c], s);
        atomicAdd(&stats[Hdim + c], sq);
    }
}

__global__ void bnfin_kernel(const float* __restrict__ stats, const float* __restrict__ g,
                             const float* __restrict__ b, float* __restrict__ ss, float invN) {
    int c = threadIdx.x;
    float mean = stats[c] * invN;
    float var  = stats[Hdim + c] * invN - mean * mean;   // biased var, as jnp.var
    float sc   = g[c] * rsqrtf(var + EPSv);
    ss[c]        = sc;
    ss[Hdim + c] = b[c] - mean * sc;
}

// fused: t = (ss? y*sc+sh : y); t = coef*t + addend; then {none | spike | conv-LIF}
__global__ __launch_bounds__(256)
void apply_kernel(const float* __restrict__ y, const float* __restrict__ ss,
                  float coef, const float* __restrict__ addend,
                  float* __restrict__ vconv, int mode, float th,
                  float* __restrict__ out, int nrows) {
    int tid = blockIdx.x * blockDim.x + threadIdx.x;
    if (tid >= nrows * (Hdim / 4)) return;
    const int c = (tid & 15) * 4;
    const unsigned off = (unsigned)tid * 4;
    float4 t = *(const float4*)(y + off);
    if (ss) {
        t.x = t.x * ss[c + 0] + ss[Hdim + c + 0];
        t.y = t.y * ss[c + 1] + ss[Hdim + c + 1];
        t.z = t.z * ss[c + 2] + ss[Hdim + c + 2];
        t.w = t.w * ss[c + 3] + ss[Hdim + c + 3];
    }
    if (coef != 1.0f) { t.x *= coef; t.y *= coef; t.z *= coef; t.w *= coef; }
    if (addend) {
        float4 a = *(const float4*)(addend + off);
        t.x += a.x; t.y += a.y; t.z += a.z; t.w += a.w;
    }
    if (mode == 1) {                                   // fresh-state LIF -> spike
        t.x = (t.x >= th) ? 1.f : 0.f;
        t.y = (t.y >= th) ? 1.f : 0.f;
        t.z = (t.z >= th) ? 1.f : 0.f;
        t.w = (t.w >= th) ? 1.f : 0.f;
    } else if (mode == 2) {                            // stateful conv-LIF, hard reset
        float4 v = *(const float4*)(vconv + off);
        v.x += t.x; v.y += t.y; v.z += t.z; v.w += t.w;
        float s0 = (v.x >= th) ? 1.f : 0.f;
        float s1 = (v.y >= th) ? 1.f : 0.f;
        float s2 = (v.z >= th) ? 1.f : 0.f;
        float s3 = (v.w >= th) ? 1.f : 0.f;
        v.x *= (1.f - s0); v.y *= (1.f - s1); v.z *= (1.f - s2); v.w *= (1.f - s3);
        *(float4*)(vconv + off) = v;
        t.x = s0; t.y = s1; t.z = s2; t.w = s3;
    }
    *(float4*)(out + off) = t;
}

// fused attention: per row, qk = spike(mean(q*k)); out = qk * v
// 16 lanes per row, butterfly reduce leaves the full dot in every lane.
__global__ __launch_bounds__(256)
void att_fused_kernel(const float* __restrict__ q, const float* __restrict__ k,
                      const float* __restrict__ v, float* __restrict__ out, int nrows) {
    int tid = blockIdx.x * blockDim.x + threadIdx.x;
    if ((tid >> 4) >= nrows) return;                 // uniform per 16-lane group
    const unsigned off = (unsigned)tid * 4;
    float4 qv = *(const float4*)(q + off);
    float4 kv = *(const float4*)(k + off);
    float p = qv.x * kv.x + qv.y * kv.y + qv.z * kv.z + qv.w * kv.w;
    p += __shfl_xor(p, 1);
    p += __shfl_xor(p, 2);
    p += __shfl_xor(p, 4);
    p += __shfl_xor(p, 8);
    const float s = (p * (1.0f / Hdim) >= ATT_THv) ? 1.f : 0.f;
    float4 vv = *(const float4*)(v + off);
    vv.x *= s; vv.y *= s; vv.z *= s; vv.w *= s;
    *(float4*)(out + off) = vv;
}

__global__ void wvec_kernel(const float* __restrict__ Wlin, float* __restrict__ wvec) {
    int c = threadIdx.x;
    if (c >= Hdim) return;
    float s = 0.f;
    for (int j = 0; j < Hdim; ++j) s += Wlin[c * Hdim + j];
    wvec[c] = s;
}

// z[n] = s[n,:] . wvec   (decoder collapsed: (s@W).sum(1) == s @ rowsum(W))
__global__ __launch_bounds__(256)
void z_kernel(const float* __restrict__ s, const float* __restrict__ wvec,
              float* __restrict__ z, int nrows) {
    int tid = blockIdx.x * blockDim.x + threadIdx.x;
    const int row = tid >> 4, part = tid & 15;
    if (row >= nrows) return;
    float4 a = *(const float4*)(s + (unsigned)tid * 4);
    float4 b = *(const float4*)(wvec + part * 4);
    float p = a.x * b.x + a.y * b.y + a.z * b.z + a.w * b.w;
    p += __shfl_xor(p, 1);
    p += __shfl_xor(p, 2);
    p += __shfl_xor(p, 4);
    p += __shfl_xor(p, 8);
    if (part == 0) z[row] = p;
}

// ------------------------------------------------------------------- driver
extern "C" void kernel_launch(void* const* d_in, const int* in_sizes, int n_in,
                              void* d_out, int out_size, void* d_ws, size_t ws_size,
                              hipStream_t stream) {
    (void)n_in; (void)out_size; (void)ws_size;
    const float* x     = (const float*)d_in[0];
    const int*   eidx  = (const int*)d_in[1];
    const int*   emask = (const int*)d_in[2];
    const float* Wg    = (const float*)d_in[3];
    const float* bg    = (const float*)d_in[4];
    const float* bng   = (const float*)d_in[5];
    const float* bnb   = (const float*)d_in[6];
    const float* Wlin  = (const float*)d_in[7];
    const int N = in_sizes[0] / Hdim;
    const int E = in_sizes[2];
    const int* src = eidx;
    const int* dst = eidx + E;
    float* out = (float*)d_out;

    const size_t NH = (size_t)N * Hdim;
    float* p = (float*)d_ws;
    float* TMP   = p; p += NH;       // GEMM output (xW)
    float* AGG   = p; p += NH;       // GCN accumulator / in-place results
    float* X2    = p; p += NH;
    float* X3    = p; p += NH;
    float* XATT  = p; p += NH;
    float* Q     = p; p += NH;
    float* K     = p; p += NH;
    float* V     = p; p += NH;
    float* VCONV = p; p += NH;
    float* DIS1  = p; p += N;
    float* DIS2  = p; p += N;
    float* WVEC  = p; p += Hdim;
    float* STATS = p; p += 2 * Hdim;
    float* SS    = p; p += 2 * Hdim;

    const int b_n    = (N + 255) / 256;
    const int b_e    = (E + 255) / 256;
    const int b_nh   = (int)((NH + 255) / 256);
    const int b_nh4  = (int)((NH / 4 + 255) / 256);
    const int b_gemm = (N + 127) / 128;
    const int b_edge = (int)(((long)E * 16 + 255) / 256);

    // one-time setup: degrees -> dis, conv-LIF state, decoder vector
    fill_kernel<<<b_n, 256, 0, stream>>>(DIS1, 1.0f, N);
    fill_kernel<<<b_n, 256, 0, stream>>>(DIS2, 1.0f, N);
    deg_kernel<<<b_e, 256, 0, stream>>>(DIS1, dst, nullptr, E);
    deg_kernel<<<b_e, 256, 0, stream>>>(DIS2, dst, emask, E);
    rsqrt_kernel<<<b_n, 256, 0, stream>>>(DIS1, N);
    rsqrt_kernel<<<b_n, 256, 0, stream>>>(DIS2, N);
    fill_kernel<<<b_nh, 256, 0, stream>>>(VCONV, 0.0f, (int)NH);
    wvec_kernel<<<1, 64, 0, stream>>>(Wlin, WVEC);

    auto run_gcn = [&](const float* Ain, int spikeA, int wi, int bni,
                       const float* disv, const int* maskv) {
        gemm_wmma_kernel<<<b_gemm, 256, 0, stream>>>(Ain, Wg + (size_t)wi * Hdim * Hdim,
                                                     TMP, spikeA, N);
        agg_init_kernel<<<b_nh4, 256, 0, stream>>>(TMP, disv, bg + wi * Hdim, AGG, N);
        edge_agg_kernel<<<b_edge, 256, 0, stream>>>(TMP, AGG, src, dst, maskv, disv, E);
        zero_stats_kernel<<<1, 128, 0, stream>>>(STATS);
        stats_kernel<<<512, 256, 0, stream>>>(AGG, STATS, N);
        bnfin_kernel<<<1, 64, 0, stream>>>(STATS, bng + bni * Hdim, bnb + bni * Hdim,
                                           SS, 1.0f / (float)N);
    };
    auto run_bn = [&](const float* xin, int bni) {
        zero_stats_kernel<<<1, 128, 0, stream>>>(STATS);
        stats_kernel<<<512, 256, 0, stream>>>(xin, STATS, N);
        bnfin_kernel<<<1, 64, 0, stream>>>(STATS, bng + bni * Hdim, bnb + bni * Hdim,
                                           SS, 1.0f / (float)N);
    };
    auto run_apply = [&](const float* y, const float* ssp, float coef, const float* add,
                         float* vc, int mode, float th, float* outb) {
        apply_kernel<<<b_nh4, 256, 0, stream>>>(y, ssp, coef, add, vc, mode, th, outb, N);
    };

    auto run_enc = [&](const float* disv, const int* maskv, float* zout) {
        // 1: bn0(gcn0(x)) -> conv_lif (stateful)          [x1 kept in AGG]
        run_gcn(x, 0, 0, 0, disv, maskv);
        run_apply(AGG, SS, 1.f, nullptr, VCONV, 2, THv, AGG);
        // 2: x2 = bn1(gcn1(x1))
        run_gcn(AGG, 0, 1, 1, disv, maskv);
        run_apply(AGG, SS, 1.f, nullptr, nullptr, 0, 0.f, X2);
        // 3: positional branch: spike(x2) fused into GEMM A-load
        run_gcn(X2, 1, 2, 2, disv, maskv);
        run_apply(AGG, SS, 1.f, X2, nullptr, 0, 0.f, AGG);   // bn2(gcn2)+x2
        run_bn(AGG, 3);
        run_apply(AGG, SS, 1.f, nullptr, nullptr, 0, 0.f, X3); // x3 = bn3(...)
        // 4: x_att = conv_lif(x3) (stateful, identity bn)
        run_apply(X3, nullptr, 1.f, nullptr, VCONV, 2, THv, XATT);
        // 5-7: q,k,v spikes
        run_gcn(XATT, 0, 3, 4, disv, maskv);
        run_apply(AGG, SS, 1.f, nullptr, nullptr, 1, THv, Q);
        run_gcn(XATT, 0, 4, 5, disv, maskv);
        run_apply(AGG, SS, 1.f, nullptr, nullptr, 1, THv, K);
        run_gcn(XATT, 0, 5, 6, disv, maskv);
        run_apply(AGG, SS, 1.f, nullptr, nullptr, 1, THv, V);
        // 8: fused attention gate: att = spike(mean(q*k)) * v  (into Q)
        att_fused_kernel<<<b_nh4, 256, 0, stream>>>(Q, K, V, Q, N);
        // 9: out = bn8(x3 + 0.125 * bn7(gcn6(att)))   [result into X2]
        run_gcn(Q, 0, 6, 7, disv, maskv);
        run_apply(AGG, SS, 0.125f, X3, nullptr, 0, 0.f, AGG);
        run_bn(AGG, 8);
        run_apply(AGG, SS, 1.f, nullptr, nullptr, 0, 0.f, X2);
        // decoder
        z_kernel<<<b_nh4, 256, 0, stream>>>(X2, WVEC, zout, N);
    };

    run_enc(DIS1, nullptr, out);        // enc pass 1, w = 1
    run_enc(DIS2, emask, out + N);      // enc pass 2, w = mask (VCONV carries over)
}
